// BalancedIPRMPNNModel_89876485636291
// MI455X (gfx1250) — compile-verified, matrix operand order checked by reading, not run on hardware
//
#include <hip/hip_runtime.h>
#include <hip/hip_bf16.h>
#include <stddef.h>
#include <stdint.h>

// ---------------------------------------------------------------------------
// Problem constants (match reference)
// ---------------------------------------------------------------------------
#define GG   128   // graphs
#define NPG  512   // nodes per graph
#define HH   128   // hidden
#define NN   (GG * NPG)
#define OUTF 10
#define VV   64
#define EPG  8192  // edges per graph

typedef __bf16 bf16_t;
typedef __attribute__((ext_vector_type(16))) __bf16 v16bf;
typedef __attribute__((ext_vector_type(8)))  __bf16 v8bf;
typedef __attribute__((ext_vector_type(8)))  float  v8f;

// LDS tile pitch in bf16 elements (row = 272B -> consecutive rows start on
// different banks, avoiding 16-way conflicts on 16B fragment reads)
#define PITCH 136

// ---------------------------------------------------------------------------
// Async Global->LDS copy (CDNA5): per-lane 16B, tracked by ASYNCcnt.
// lds_off is the workgroup-relative LDS byte offset (low 32 bits of the
// generic-aperture LDS address).
// ---------------------------------------------------------------------------
__device__ __forceinline__ void async_ld_b128(uint32_t lds_off, const void* g) {
  asm volatile("global_load_async_to_lds_b128 %0, %1, off"
               :: "v"(lds_off), "v"(g) : "memory");
}
__device__ __forceinline__ void wait_async0() {
  asm volatile("s_wait_asynccnt 0" ::: "memory");
}
__device__ __forceinline__ uint32_t lds_off_of(const void* p) {
  return (uint32_t)(uintptr_t)p;   // flat LDS aperture: addr[31:0] = LDS offset
}

// ---------------------------------------------------------------------------
// WMMA helpers (v_wmma_f32_16x16x32_bf16, wave32)
// A fragment: lane L<16 -> row L, K = {kb..kb+7, kb+16..kb+23}
//             lane 16+L -> row L, K = {kb+8..kb+15, kb+24..kb+31}
// B fragment: identical indexing on a [N][K] (transposed-B) tile.
// ---------------------------------------------------------------------------
__device__ __forceinline__ v16bf frag_ld(const bf16_t* __restrict__ base,
                                         int kb, int lane) {
  const int r    = lane & 15;
  const int koff = (lane >> 4) << 3;           // 0 or 8
  const bf16_t* p = base + r * PITCH + kb + koff;
  v8bf lo = *(const v8bf*)(p);
  v8bf hi = *(const v8bf*)(p + 16);
  v16bf v;
#pragma unroll
  for (int i = 0; i < 8; ++i) { v[i] = lo[i]; v[i + 8] = hi[i]; }
  return v;
}

__device__ __forceinline__ v8f wmma_bf16(v16bf a, v16bf b, v8f c) {
  // (neg_a, A, neg_b, B, c_mod, C, reuse_a, reuse_b)
  return __builtin_amdgcn_wmma_f32_16x16x32_bf16(false, a, false, b,
                                                 (short)0, c, false, false);
}

// ---------------------------------------------------------------------------
// Kernel 1: fold W_emb@W_gcn -> WcT (bf16, stored [out][in] i.e. B-transposed),
// bc = b_emb@W_gcn (f32), and transpose+cast vW1/vW2/mW1 to bf16 [N][K].
// ---------------------------------------------------------------------------
__global__ __launch_bounds__(256) void k_prep(
    const float* __restrict__ W_emb, const float* __restrict__ b_emb,
    const float* __restrict__ W_gcn,
    const float* __restrict__ vW1, const float* __restrict__ vW2,
    const float* __restrict__ mW1,
    bf16_t* __restrict__ WcT, float* __restrict__ bc,
    bf16_t* __restrict__ vW1T, bf16_t* __restrict__ vW2T,
    bf16_t* __restrict__ mW1T) {
  const int tid = threadIdx.x;
  for (int idx = tid; idx < HH * HH; idx += 256) {
    const int i = idx >> 7, j = idx & 127;
    float s = 0.f;
#pragma unroll 8
    for (int k = 0; k < HH; ++k) s = fmaf(W_emb[i * HH + k], W_gcn[k * HH + j], s);
    WcT[j * HH + i] = (bf16_t)s;
  }
  if (tid < HH) {
    float s = 0.f;
    for (int k = 0; k < HH; ++k) s = fmaf(b_emb[k], W_gcn[k * HH + tid], s);
    bc[tid] = s;
  }
  for (int idx = tid; idx < HH * HH; idx += 256) {
    const int i = idx >> 7, j = idx & 127;
    vW1T[j * HH + i] = (bf16_t)vW1[idx];
    vW2T[j * HH + i] = (bf16_t)vW2[idx];
    mW1T[j * HH + i] = (bf16_t)mW1[idx];
  }
}

// ---------------------------------------------------------------------------
// Kernel 2: hW = x @ Wc + bc   (M=65536, N=128, K=128), bf16 WMMA, f32 accum.
// One block = 128 rows x full N=128. 8 waves, one 16-row M-tile each,
// 8 accumulators (8 N-tiles) per wave. B staged by the async-to-LDS engine.
// ---------------------------------------------------------------------------
__global__ __launch_bounds__(256) void k_hw_gemm(
    const float* __restrict__ x, const bf16_t* __restrict__ WcT,
    const float* __restrict__ bc, float* __restrict__ hW) {
  extern __shared__ char smem[];
  bf16_t* As = (bf16_t*)smem;                       // [128][PITCH]
  bf16_t* Bs = (bf16_t*)(smem + 128 * PITCH * 2);   // [128][PITCH]
  const int tid  = threadIdx.x;
  const int row0 = blockIdx.x * 128;

  // async-stage B (bf16 [N][K], contiguous rows -> pitched LDS rows)
  {
    const uint32_t bs0 = lds_off_of(Bs);
    for (int c = tid; c < 128 * 16; c += 256) {       // 2048 x 16B = 32KB
      const int row = c >> 4, part = c & 15;
      async_ld_b128(bs0 + (uint32_t)(row * (PITCH * 2) + part * 16),
                    WcT + row * HH + part * 8);
    }
  }

  // prefetch next M-tile of x into caches
  if (tid < 128 && blockIdx.x + 1 < gridDim.x)
    __builtin_prefetch(x + (size_t)(row0 + 128 + tid) * HH, 0, 3);

  // stage A (f32 -> bf16 conversion must pass through VGPRs)
  for (int idx = tid; idx < 128 * 32; idx += 256) {   // float4 granularity
    const int r = idx >> 5, c4 = idx & 31;
    float4 v = ((const float4*)(x + (size_t)(row0 + r) * HH))[c4];
    bf16_t* d = As + r * PITCH + c4 * 4;
    d[0] = (bf16_t)v.x; d[1] = (bf16_t)v.y; d[2] = (bf16_t)v.z; d[3] = (bf16_t)v.w;
  }
  wait_async0();
  __syncthreads();

  const int lane = tid & 31, wid = tid >> 5;
  const int m0 = wid * 16;
  v8f acc[8] = {};
#pragma unroll
  for (int kb = 0; kb < HH; kb += 32) {
    v16bf a = frag_ld(As + m0 * PITCH, kb, lane);
#pragma unroll
    for (int j = 0; j < 8; ++j) {
      v16bf b = frag_ld(Bs + j * 16 * PITCH, kb, lane);
      acc[j] = wmma_bf16(a, b, acc[j]);
    }
  }
  const int cr = lane & 15;
  const int rb = m0 + ((lane >> 4) << 3);
#pragma unroll
  for (int j = 0; j < 8; ++j) {
    const int col = j * 16 + cr;
    const float bias = bc[col];
#pragma unroll
    for (int v = 0; v < 8; ++v)
      hW[(size_t)(row0 + rb + v) * HH + col] = acc[j][v] + bias;
  }
}

// ---------------------------------------------------------------------------
// Kernel 3: per-graph GCN aggregation, fully in LDS (one block per graph).
//   deg/dinv in LDS; 4 feature-chunk passes of 32 features:
//   raw chunk async-copied Global->LDS, then s[n][f] = dinv[n]*raw (self-loop
//   init of acc), acc[dst] += s[src] via ds_add_f32, then
//   h[n][f] = relu(dinv[n]*acc[n][f] + b_gcn[f])   (in-place over hW)
// ---------------------------------------------------------------------------
__global__ __launch_bounds__(512) void k_gcn_agg(
    const int* __restrict__ src, const int* __restrict__ dst,
    const float* __restrict__ b_gcn, float* __restrict__ h) {
  extern __shared__ char smem[];
  float* deg  = (float*)smem;             // [512]
  float* dinv = deg + NPG;                // [512]
  float* sva  = dinv + NPG;               // [512][32]
  float* acc  = sva + NPG * 32;           // [512][32]
  const int g = blockIdx.x, tid = threadIdx.x;
  const int ebase = g * EPG;
  const uint32_t acc0 = lds_off_of(acc);

  if (tid < NPG) deg[tid] = 1.0f;          // self loop
  __syncthreads();
  for (int e = tid; e < EPG; e += 512)
    atomicAdd(&deg[dst[ebase + e] & (NPG - 1)], 1.0f);
  __syncthreads();
  if (tid < NPG) dinv[tid] = rsqrtf(deg[tid]);

  for (int fb = 0; fb < HH; fb += 32) {
    // async copy raw h chunk (512 nodes x 32 floats, 128B/node) into acc
    for (int c = tid; c < NPG * 8; c += 512) {        // 4096 x 16B = 64KB
      const int n = c >> 3, part = c & 7;
      async_ld_b128(acc0 + (uint32_t)c * 16,
                    h + (size_t)(g * NPG + n) * HH + fb + part * 4);
    }
    wait_async0();
    __syncthreads();
    // scale by dinv: sva = acc = dinv[n] * raw  (self-loop contribution)
    for (int idx = tid; idx < NPG * 32; idx += 512) {
      const int n = idx >> 5;
      const float v = dinv[n] * acc[idx];
      sva[idx] = v;
      acc[idx] = v;
    }
    __syncthreads();
    // edge scatter: vector ds reads + ds_add_f32 atomics
    for (int e = tid; e < EPG; e += 512) {
      const int sl = src[ebase + e] & (NPG - 1);
      const int dl = dst[ebase + e] & (NPG - 1);
      const float4* sp4 = (const float4*)(sva + sl * 32);
      float* ap = acc + dl * 32;
#pragma unroll
      for (int q = 0; q < 8; ++q) {
        const float4 vv = sp4[q];
        atomicAdd(&ap[q * 4 + 0], vv.x);
        atomicAdd(&ap[q * 4 + 1], vv.y);
        atomicAdd(&ap[q * 4 + 2], vv.z);
        atomicAdd(&ap[q * 4 + 3], vv.w);
      }
    }
    __syncthreads();
    for (int idx = tid; idx < NPG * 32; idx += 512) {
      const int n = idx >> 5, f = idx & 31;
      const float o = dinv[n] * acc[idx] + b_gcn[fb + f];
      h[(size_t)(g * NPG + n) * HH + fb + f] = fmaxf(o, 0.f);
    }
    __syncthreads();
  }
}

// ---------------------------------------------------------------------------
// Kernel 4: per-graph pooling + virtual-node MLP + mean over V.
//   virt = ew^T(64x512) @ h(512x128)      (WMMA, K chunked by 128)
//   z1   = relu(virt@vW1+vb1); z2 = z1@vW2+vb2; gf = mean_v z2
// 8 waves: wave w -> M-tile w/2, N-tiles (w&1)*4 .. +4
// ---------------------------------------------------------------------------
__global__ __launch_bounds__(256) void k_pool_mlp(
    const float* __restrict__ hrelu, const float* __restrict__ ew,
    const bf16_t* __restrict__ vW1T, const float* __restrict__ vb1,
    const bf16_t* __restrict__ vW2T, const float* __restrict__ vb2,
    float* __restrict__ gf) {
  extern __shared__ char smem[];
  bf16_t* Abuf = (bf16_t*)smem;                                 // 64 x PITCH
  bf16_t* Bbuf = (bf16_t*)(smem + 64 * PITCH * 2);              // 128 x PITCH
  bf16_t* Zbuf = (bf16_t*)(smem + (64 + 128) * PITCH * 2);      // 64 x PITCH
  float*  gfb  = (float*)(smem + (64 + 128 + 64) * PITCH * 2);  // 128

  const int g = blockIdx.x, tid = threadIdx.x;
  const int lane = tid & 31, wid = tid >> 5;
  const int mt  = wid >> 1;           // M-tile 0..3 (rows of V)
  const int nt0 = (wid & 1) * 4;      // N-tile base
  const int cr  = lane & 15;
  const int rb  = mt * 16 + ((lane >> 4) << 3);
  if (tid < HH) gfb[tid] = 0.f;

  // ---- pooling GEMM, K=512 in 4 chunks of 128 ----
  v8f acc[4] = {};
  for (int ck = 0; ck < 4; ++ck) {
    const int nb = g * NPG + ck * 128;
    for (int idx = tid; idx < VV * 128; idx += 256) {      // ewT chunk
      const int v = idx >> 7, kk = idx & 127;
      Abuf[v * PITCH + kk] = (bf16_t)ew[(size_t)(nb + kk) * VV + v];
    }
    for (int idx = tid; idx < HH * 128; idx += 256) {      // h^T chunk
      const int hc = idx >> 7, kk = idx & 127;
      Bbuf[hc * PITCH + kk] = (bf16_t)hrelu[(size_t)(nb + kk) * HH + hc];
    }
    __syncthreads();
#pragma unroll
    for (int kb = 0; kb < 128; kb += 32) {
      v16bf a = frag_ld(Abuf + mt * 16 * PITCH, kb, lane);
#pragma unroll
      for (int j = 0; j < 4; ++j) {
        v16bf b = frag_ld(Bbuf + (nt0 + j) * 16 * PITCH, kb, lane);
        acc[j] = wmma_bf16(a, b, acc[j]);
      }
    }
    __syncthreads();
  }
  // virt -> Abuf (bf16), stage vW1T -> Bbuf
#pragma unroll
  for (int j = 0; j < 4; ++j) {
    const int col = (nt0 + j) * 16 + cr;
#pragma unroll
    for (int v = 0; v < 8; ++v) Abuf[(rb + v) * PITCH + col] = (bf16_t)acc[j][v];
  }
  for (int idx = tid; idx < HH * HH; idx += 256)
    Bbuf[(idx >> 7) * PITCH + (idx & 127)] = vW1T[idx];
  __syncthreads();

  // ---- z1 = relu(virt@vW1 + vb1) ----
  v8f a2[4] = {};
#pragma unroll
  for (int kb = 0; kb < HH; kb += 32) {
    v16bf a = frag_ld(Abuf + mt * 16 * PITCH, kb, lane);
#pragma unroll
    for (int j = 0; j < 4; ++j) {
      v16bf b = frag_ld(Bbuf + (nt0 + j) * 16 * PITCH, kb, lane);
      a2[j] = wmma_bf16(a, b, a2[j]);
    }
  }
#pragma unroll
  for (int j = 0; j < 4; ++j) {
    const int col = (nt0 + j) * 16 + cr;
    const float bias = vb1[col];
#pragma unroll
    for (int v = 0; v < 8; ++v)
      Zbuf[(rb + v) * PITCH + col] = (bf16_t)fmaxf(a2[j][v] + bias, 0.f);
  }
  __syncthreads();
  for (int idx = tid; idx < HH * HH; idx += 256)
    Bbuf[(idx >> 7) * PITCH + (idx & 127)] = vW2T[idx];
  __syncthreads();

  // ---- z2 = z1@vW2 + vb2; accumulate mean over V ----
  v8f a3[4] = {};
#pragma unroll
  for (int kb = 0; kb < HH; kb += 32) {
    v16bf a = frag_ld(Zbuf + mt * 16 * PITCH, kb, lane);
#pragma unroll
    for (int j = 0; j < 4; ++j) {
      v16bf b = frag_ld(Bbuf + (nt0 + j) * 16 * PITCH, kb, lane);
      a3[j] = wmma_bf16(a, b, a3[j]);
    }
  }
#pragma unroll
  for (int j = 0; j < 4; ++j) {
    const int col = (nt0 + j) * 16 + cr;
    const float bias = vb2[col];
    float s8 = 0.f;
#pragma unroll
    for (int v = 0; v < 8; ++v) s8 += a3[j][v] + bias;
    atomicAdd(&gfb[col], s8);
  }
  __syncthreads();
  if (tid < HH) gf[(size_t)g * HH + tid] = gfb[tid] * (1.0f / (float)VV);
}

// ---------------------------------------------------------------------------
// Kernel 5: final MLP.  t = relu(gf@mW1+mb1) via WMMA; out = t@mW2+mb2 (N=10)
// ---------------------------------------------------------------------------
__global__ __launch_bounds__(256) void k_final(
    const float* __restrict__ gf, const bf16_t* __restrict__ mW1T,
    const float* __restrict__ mb1, const float* __restrict__ mW2,
    const float* __restrict__ mb2, float* __restrict__ out) {
  extern __shared__ char smem[];
  bf16_t* As = (bf16_t*)smem;                           // 128 x PITCH
  bf16_t* Bs = (bf16_t*)(smem + 128 * PITCH * 2);       // 128 x PITCH
  bf16_t* Ts = (bf16_t*)(smem + 256 * PITCH * 2);       // 128 x PITCH
  const int tid = threadIdx.x, lane = tid & 31, wid = tid >> 5;

  for (int idx = tid; idx < HH * HH; idx += 256) {
    const int r = idx >> 7, c = idx & 127;
    As[r * PITCH + c] = (bf16_t)gf[idx];
    Bs[r * PITCH + c] = mW1T[idx];
  }
  __syncthreads();

  const int m0 = wid * 16;
  v8f acc[8] = {};
#pragma unroll
  for (int kb = 0; kb < HH; kb += 32) {
    v16bf a = frag_ld(As + m0 * PITCH, kb, lane);
#pragma unroll
    for (int j = 0; j < 8; ++j) {
      v16bf b = frag_ld(Bs + j * 16 * PITCH, kb, lane);
      acc[j] = wmma_bf16(a, b, acc[j]);
    }
  }
  const int cr = lane & 15;
  const int rb = m0 + ((lane >> 4) << 3);
#pragma unroll
  for (int j = 0; j < 8; ++j) {
    const int col = j * 16 + cr;
    const float bias = mb1[col];
#pragma unroll
    for (int v = 0; v < 8; ++v)
      Ts[(rb + v) * PITCH + col] = (bf16_t)fmaxf(acc[j][v] + bias, 0.f);
  }
  __syncthreads();
  for (int idx = tid; idx < GG * OUTF; idx += 256) {
    const int gr = idx / OUTF, o = idx % OUTF;
    float s = mb2[o];
#pragma unroll 8
    for (int k = 0; k < HH; ++k)
      s = fmaf((float)Ts[gr * PITCH + k], mW2[k * OUTF + o], s);
    out[idx] = s;
  }
}

// ---------------------------------------------------------------------------
// Launch
// ---------------------------------------------------------------------------
extern "C" void kernel_launch(void* const* d_in, const int* in_sizes, int n_in,
                              void* d_out, int out_size, void* d_ws, size_t ws_size,
                              hipStream_t stream) {
  const float* x       = (const float*)d_in[0];
  const int*   eidx    = (const int*)d_in[1];
  // d_in[2] = batch (unused: graphs are contiguous, equal size)
  const float* W_emb   = (const float*)d_in[3];
  const float* b_emb   = (const float*)d_in[4];
  const float* W_gcn   = (const float*)d_in[5];
  const float* b_gcn   = (const float*)d_in[6];
  const float* ew      = (const float*)d_in[7];
  const float* vW1     = (const float*)d_in[8];
  const float* vb1     = (const float*)d_in[9];
  const float* vW2     = (const float*)d_in[10];
  const float* vb2     = (const float*)d_in[11];
  const float* mW1     = (const float*)d_in[12];
  const float* mb1     = (const float*)d_in[13];
  const float* mW2     = (const float*)d_in[14];
  const float* mb2     = (const float*)d_in[15];
  float* out = (float*)d_out;

  const int E = in_sizes[1] / 2;
  const int* src = eidx;
  const int* dst = eidx + E;

  // workspace layout
  char* ws = (char*)d_ws;
  float*  hW   = (float*)ws;                                  // N*H f32 (in-place -> relu'd h)
  size_t  off  = (size_t)NN * HH * sizeof(float);
  float*  gf   = (float*)(ws + off);  off += (size_t)GG * HH * sizeof(float);
  float*  bc   = (float*)(ws + off);  off += 512;
  bf16_t* WcT  = (bf16_t*)(ws + off); off += (size_t)HH * HH * sizeof(bf16_t);
  bf16_t* vW1T = (bf16_t*)(ws + off); off += (size_t)HH * HH * sizeof(bf16_t);
  bf16_t* vW2T = (bf16_t*)(ws + off); off += (size_t)HH * HH * sizeof(bf16_t);
  bf16_t* mW1T = (bf16_t*)(ws + off); off += (size_t)HH * HH * sizeof(bf16_t);
  (void)ws_size; (void)n_in; (void)out_size; (void)E;

  // 1) fold linear layers, transpose/cast weights to bf16
  k_prep<<<1, 256, 0, stream>>>(W_emb, b_emb, W_gcn, vW1, vW2, mW1,
                                WcT, bc, vW1T, vW2T, mW1T);

  // 2) hW = x @ Wc + bc   (WMMA + async-to-LDS staging)
  k_hw_gemm<<<NN / 128, 256, 2 * 128 * PITCH * 2, stream>>>(x, WcT, bc, hW);

  // 3) per-graph GCN aggregation in LDS (async-to-LDS input staging)
  k_gcn_agg<<<GG, 512, (NPG * 2 + NPG * 32 * 2) * sizeof(float), stream>>>(
      src, dst, b_gcn, hW);

  // 4) per-graph pooling + virtual-node MLP + mean  (WMMA)
  k_pool_mlp<<<GG, 256, (64 + 128 + 64) * PITCH * 2 + 128 * sizeof(float),
               stream>>>(hW, ew, vW1T, vb1, vW2T, vb2, gf);

  // 5) final MLP  (WMMA + tail)
  k_final<<<1, 256, 3 * 128 * PITCH * 2, stream>>>(gf, mW1T, mb1, mW2, mb2, out);
}
